// MultiHeadAttention_8074538516995
// MI455X (gfx1250) — matrix operand order repaired and checked by reading, not compiled, not run on hardware
//
#include <hip/hip_runtime.h>
#include <hip/hip_bf16.h>
#include <stdint.h>

typedef _Float16 v16h __attribute__((ext_vector_type(16)));
typedef _Float16 v8h  __attribute__((ext_vector_type(8)));
typedef float    v8f  __attribute__((ext_vector_type(8)));
typedef float    v4f  __attribute__((ext_vector_type(4)));
typedef unsigned int u32x4 __attribute__((ext_vector_type(4)));
typedef int      i32x8 __attribute__((ext_vector_type(8)));
typedef int      i32x4 __attribute__((ext_vector_type(4)));

#define B_  2
#define S_  2048
#define H_  1024
#define NH_ 16
#define DK_ 64

#if __has_builtin(__builtin_amdgcn_tensor_load_to_lds)
#define HAVE_TDM 1
#else
#define HAVE_TDM 0
#endif

// ---------- CDNA5 async copy: global -> LDS, 16 bytes per lane ----------
__device__ __forceinline__ void async_cp16(void* lds, const void* g) {
  asm volatile("global_load_async_to_lds_b128 %0, %1, off"
               :: "v"((uint32_t)(uintptr_t)lds),
                  "v"((unsigned long long)(uintptr_t)g)
               : "memory");
}
__device__ __forceinline__ void wait_async_0() {
  asm volatile("s_wait_asynccnt 0" ::: "memory");
}
__device__ __forceinline__ void wait_async_2() {
  asm volatile("s_wait_asynccnt 2" ::: "memory");
}

#if HAVE_TDM
// ---------- CDNA5 Tensor Data Mover: 2D f16 tile, global -> LDS ----------
// D# layout per CDNA5 ISA ch.8: group0 = {count/type/addrs}, group1 = dims.
__device__ __forceinline__ void tdm_load_2d_f16(void* lds, const void* g,
                                                unsigned tile_d0, unsigned tile_d1,
                                                unsigned tensor_d0, unsigned tensor_d1,
                                                unsigned stride0) {
  const unsigned long long ga = (unsigned long long)(uintptr_t)g;
  u32x4 g0;
  g0[0] = 1u;                                   // count=1 (valid), user mode
  g0[1] = (unsigned)(uintptr_t)lds;             // lds_addr (bytes)
  g0[2] = (unsigned)(ga & 0xFFFFFFFFu);         // global_addr[31:0]
  g0[3] = (unsigned)((ga >> 32) & 0x1FFFFFFu)   // global_addr[56:32]
          | (2u << 30);                         // type = 2 ("image")
  i32x8 g1;
  g1[0] = (int)(1u << 16);                      // data_size = 1 -> 2 bytes
  g1[1] = (int)((tensor_d0 & 0xFFFFu) << 16);   // tensor_dim0[15:0] @ [63:48]
  g1[2] = (int)((tensor_d0 >> 16) |
                ((tensor_d1 & 0xFFFFu) << 16)); // t0[31:16] @ [79:64], t1[15:0] @ [95:80]
  g1[3] = (int)((tensor_d1 >> 16) |
                (tile_d0 << 16));               // t1[31:16] @ [111:96], tile_dim0 @ [127:112]
  g1[4] = (int)tile_d1;                         // tile_dim1 @ [143:128], tile_dim2 = 0
  g1[5] = (int)stride0;                         // tensor_dim0_stride[31:0]
  g1[6] = 0;                                    // stride0 hi, dim1_stride lo
  g1[7] = 0;                                    // dim1_stride hi
  const i32x4 z4 = {0, 0, 0, 0};                // groups 2/3 unused (2D)
  const i32x8 z8 = {0, 0, 0, 0, 0, 0, 0, 0};
  __builtin_amdgcn_tensor_load_to_lds(g0, g1, z4, z4, z8, 0);
}
#endif

// ---------- WMMA fragment loaders (wave32, 16x16x32 f16) ----------
// A fragment: 16 rows x 32 K. Lane l: row = l&15; lanes 0-15 hold K 0-7,16-23;
// lanes 16-31 hold K 8-15,24-31 (per CDNA5 ISA A-matrix table).
__device__ __forceinline__ v16h lds_load_a16(const _Float16* base, int ldk) {
  const int lane = threadIdx.x & 31;
  const _Float16* p = base + (lane & 15) * ldk + ((lane >> 4) * 8);
  union { v16h v; v8h h[2]; } u;
  u.h[0] = *(const v8h*)(p);
  u.h[1] = *(const v8h*)(p + 16);
  return u.v;
}
// B fragment from row-major [N][K] tile (i.e. B^T rows K-contiguous).
// Lane l: column n = l&15; lanes 0-15 hold K 0-7,16-23; lanes 16-31 K 8-15,24-31.
__device__ __forceinline__ v16h lds_load_b16(const _Float16* base, int ldk) {
  const int lane = threadIdx.x & 31;
  const _Float16* p = base + (lane & 15) * ldk + ((lane >> 4) * 8);
  union { v16h v; v8h h[2]; } u;
  u.h[0] = *(const v8h*)(p);
  u.h[1] = *(const v8h*)(p + 16);
  return u.v;
}

__device__ __forceinline__ v8f wmma_f16(v16h a, v16h b, v8f c) {
  return __builtin_amdgcn_wmma_f32_16x16x32_f16(false, a, false, b, (short)0, c,
                                                false, false);
}

// =================================================================
// Kernel 0: one-shot fp32 -> f16 conversion (streaming, 8 elts/thread)
// =================================================================
__global__ __launch_bounds__(256)
void cvt_f16_kernel(const float* __restrict__ src, _Float16* __restrict__ dst,
                    int n) {
  const int i = (blockIdx.x * 256 + threadIdx.x) * 8;
  if (i >= n) return;
  v4f f0 = *(const v4f*)(src + i);
  v4f f1 = *(const v4f*)(src + i + 4);
  v8h h;
#pragma unroll
  for (int j = 0; j < 4; ++j) { h[j] = (_Float16)f0[j]; h[j + 4] = (_Float16)f1[j]; }
  *(v8h*)(dst + i) = h;
}

// =================================================================
// Kernel 1: fused QKV projection (f16 operands).  out = x @ W^T + bias,
// written f16 in [B, NH, S, DK].  blockIdx.z selects Q/K/V.
// Block tile 128x128, K-step 32, 256 threads = 8 waves (4x2), each wave
// computes 32x64 (8 accumulators).  Double-buffered staging:
//   A tile -> GLOBAL_LOAD_ASYNC_TO_LDS_B128 (ASYNCcnt)
//   B tile -> TENSOR_LOAD_TO_LDS issued by wave 0 (TENSORcnt)
// =================================================================
__global__ __launch_bounds__(256)
void qkv_proj_kernel(const _Float16* __restrict__ Qh, const _Float16* __restrict__ Kh,
                     const _Float16* __restrict__ Vh,
                     const _Float16* __restrict__ Wqh, const _Float16* __restrict__ Wkh,
                     const _Float16* __restrict__ Wvh,
                     const float* __restrict__ bq, const float* __restrict__ bk,
                     const float* __restrict__ bv,
                     _Float16* __restrict__ qp, _Float16* __restrict__ kp,
                     _Float16* __restrict__ vp) {
  __shared__ _Float16 sA[2][128 * 32];
  __shared__ _Float16 sB[2][128 * 32];

  const _Float16* X; const _Float16* W; const float* bias; _Float16* Out;
  if (blockIdx.z == 0)      { X = Qh; W = Wqh; bias = bq; Out = qp; }
  else if (blockIdx.z == 1) { X = Kh; W = Wkh; bias = bk; Out = kp; }
  else                      { X = Vh; W = Wvh; bias = bv; Out = vp; }

  const int tid  = threadIdx.x;
  const int lane = tid & 31;
  const int wid  = tid >> 5;
  const int wm   = wid & 3;   // 32-row strip within 128
  const int wn   = wid >> 2;  // 64-col strip within 128
  const int m0   = blockIdx.x * 128;
  const int n0   = blockIdx.y * 128;

  const v8f zero8 = {0.f, 0.f, 0.f, 0.f, 0.f, 0.f, 0.f, 0.f};
  v8f acc[2][4];
#pragma unroll
  for (int i = 0; i < 2; ++i)
#pragma unroll
    for (int f = 0; f < 4; ++f) acc[i][f] = zero8;

  const int sr = tid >> 2;           // 0..63 staging row
  const int sc = (tid & 3) * 8;      // 0,8,16,24 staging col

  // ---- tile stager ----
  auto stage = [&](int k0, int buf) {
    async_cp16(&sA[buf][sr * 32 + sc],        X + (size_t)(m0 + sr) * H_ + k0 + sc);
    async_cp16(&sA[buf][(sr + 64) * 32 + sc], X + (size_t)(m0 + sr + 64) * H_ + k0 + sc);
#if HAVE_TDM
    if (wid == 0)
      tdm_load_2d_f16(&sB[buf][0], W + (size_t)n0 * H_ + k0,
                      /*tile*/ 32, 128, /*tensor*/ H_, H_, /*stride*/ H_);
#else
    async_cp16(&sB[buf][sr * 32 + sc],        W + (size_t)(n0 + sr) * H_ + k0 + sc);
    async_cp16(&sB[buf][(sr + 64) * 32 + sc], W + (size_t)(n0 + sr + 64) * H_ + k0 + sc);
#endif
  };

  stage(0, 0);   // prologue: tile 0 in flight

  const int NT = H_ / 32;
  for (int it = 0; it < NT; ++it) {
    const int cur = it & 1;
    const bool more = (it + 1) < NT;
    if (more) {
      stage((it + 1) * 32, cur ^ 1);   // tile it+1 in flight
#if HAVE_TDM
      wait_async_2();                                   // A of tile `it` done
      if (wid == 0) __builtin_amdgcn_s_wait_tensorcnt((short)1);  // B of tile `it` done
#else
      asm volatile("s_wait_asynccnt 4" ::: "memory");
#endif
    } else {
      wait_async_0();
#if HAVE_TDM
      if (wid == 0) __builtin_amdgcn_s_wait_tensorcnt((short)0);
#endif
    }
    __syncthreads();

    v16h a0 = lds_load_a16(&sA[cur][(wm * 32) * 32], 32);
    v16h a1 = lds_load_a16(&sA[cur][(wm * 32 + 16) * 32], 32);
#pragma unroll
    for (int f = 0; f < 4; ++f) {
      v16h b = lds_load_b16(&sB[cur][(wn * 64 + f * 16) * 32], 32);
      acc[0][f] = wmma_f16(a0, b, acc[0][f]);
      acc[1][f] = wmma_f16(a1, b, acc[1][f]);
    }
    __syncthreads();   // everyone done with buf `cur` before it is refilled
  }

  // epilogue: + bias, store f16 into [B, NH, S, DK]
  const int lhi = lane >> 4, llo = lane & 15;
#pragma unroll
  for (int fi = 0; fi < 2; ++fi) {
#pragma unroll
    for (int fj = 0; fj < 4; ++fj) {
      const int row0 = m0 + wm * 32 + fi * 16 + 8 * lhi;
      const int col  = n0 + wn * 64 + fj * 16 + llo;
      const float bvv = bias[col];
      const int h = col >> 6, d = col & 63;
#pragma unroll
      for (int r = 0; r < 8; ++r) {
        const int row = row0 + r;
        const int b = row >> 11;          // / S_
        const int s = row & (S_ - 1);
        Out[(((size_t)b * NH_ + h) * S_ + s) * DK_ + d] =
            (_Float16)(acc[fi][fj][r] + bvv);
      }
    }
  }
}

// =================================================================
// Kernel 2: flash-style masked attention per (b, h).
// Block: 128 q-rows (8 waves x 16 rows). Keys streamed in 32-wide tiles:
// K tile async-copied to LDS, V tile transposed via registers. Online softmax.
// =================================================================
__global__ __launch_bounds__(256)
void attn_kernel(const _Float16* __restrict__ qp, const _Float16* __restrict__ kp,
                 const _Float16* __restrict__ vp,
                 const unsigned char* __restrict__ mask,
                 _Float16* __restrict__ ctx) {
  __shared__ _Float16 sK[32 * 64];        // [key][d]
  __shared__ _Float16 sVt[64 * 32];       // [d][key]
  __shared__ _Float16 sP[8][16 * 32];     // per-wave P scratch [qrow][key]

  const int tid  = threadIdx.x;
  const int lane = tid & 31;
  const int wid  = tid >> 5;
  const int lhi  = lane >> 4;
  const int llo  = lane & 15;

  const int bz = blockIdx.z, hy = blockIdx.y;
  const size_t bh = (size_t)(bz * NH_ + hy) * S_ * DK_;
  const int q0 = blockIdx.x * 128 + wid * 16;

  // persistent Q A-fragments (K = d, chained 2x32)
  v16h aq[2];
  {
    const _Float16* qrow = qp + bh + (size_t)(q0 + llo) * DK_;
#pragma unroll
    for (int kd = 0; kd < 2; ++kd) {
      const _Float16* p = qrow + kd * 32 + lhi * 8;
      union { v16h v; v8h h[2]; } u;
      u.h[0] = *(const v8h*)(p);
      u.h[1] = *(const v8h*)(p + 16);
      aq[kd] = u.v;
    }
  }

  const v8f zero8 = {0.f, 0.f, 0.f, 0.f, 0.f, 0.f, 0.f, 0.f};
  v8f o[4];
  o[0] = zero8; o[1] = zero8; o[2] = zero8; o[3] = zero8;
  float mrow[8], lrow[8];
#pragma unroll
  for (int r = 0; r < 8; ++r) { mrow[r] = -3.0e38f; lrow[r] = 0.0f; }

  const int krow = tid >> 3;        // 0..31
  const int kseg = (tid & 7) * 8;   // 0..56

  for (int kb = 0; kb < S_; kb += 32) {
    // async stage K tile [32 x 64]: one 16B chunk per thread
    async_cp16(sK + tid * 8,
               kp + bh + (size_t)(kb + krow) * DK_ + kseg);
    { // stage V^T tile [64 x 32] via register transpose
      v8h v = *(const v8h*)(vp + bh + (size_t)(kb + krow) * DK_ + kseg);
#pragma unroll
      for (int j = 0; j < 8; ++j) sVt[(kseg + j) * 32 + krow] = v[j];
    }
    wait_async_0();
    __syncthreads();

    // scores S = Q K^T  (two 16-key N-tiles, chained over d)
    v8f scv[2];
    scv[0] = zero8; scv[1] = zero8;
#pragma unroll
    for (int nf = 0; nf < 2; ++nf) {
#pragma unroll
      for (int kd = 0; kd < 2; ++kd) {
        v16h b = lds_load_b16(sK + (nf * 16) * 64 + kd * 32, 64);
        scv[nf] = wmma_f16(aq[kd], b, scv[nf]);
      }
    }

    // scale, mask, online softmax (row = r + 8*lhi, col = llo across lanes)
    float p0[8], p1[8];
#pragma unroll
    for (int r = 0; r < 8; ++r) {
      const int qg = q0 + r + 8 * lhi;
      float s0 = scv[0][r] * 0.125f;
      float s1 = scv[1][r] * 0.125f;
      const size_t mbase = ((size_t)bz * S_ + qg) * S_ + kb;
      if (mask[mbase + llo])      s0 = -1.0e9f;
      if (mask[mbase + 16 + llo]) s1 = -1.0e9f;
      float mx = fmaxf(s0, s1);
      mx = fmaxf(mx, __shfl_xor(mx, 1, 32));
      mx = fmaxf(mx, __shfl_xor(mx, 2, 32));
      mx = fmaxf(mx, __shfl_xor(mx, 4, 32));
      mx = fmaxf(mx, __shfl_xor(mx, 8, 32));
      const float mn   = fmaxf(mrow[r], mx);
      const float corr = __expf(mrow[r] - mn);
      const float e0 = __expf(s0 - mn);
      const float e1 = __expf(s1 - mn);
      float rs = e0 + e1;
      rs += __shfl_xor(rs, 1, 32);
      rs += __shfl_xor(rs, 2, 32);
      rs += __shfl_xor(rs, 4, 32);
      rs += __shfl_xor(rs, 8, 32);
      lrow[r] = lrow[r] * corr + rs;
      mrow[r] = mn;
      p0[r] = e0; p1[r] = e1;
#pragma unroll
      for (int f = 0; f < 4; ++f) o[f][r] *= corr;
    }

    { // D-layout -> A-layout via per-wave LDS strip, then O += P V
      _Float16* sPw = sP[wid];
#pragma unroll
      for (int r = 0; r < 8; ++r) {
        const int row = r + 8 * lhi;
        sPw[row * 32 + llo]      = (_Float16)p0[r];
        sPw[row * 32 + 16 + llo] = (_Float16)p1[r];
      }
      v16h ap = lds_load_a16(sPw, 32);   // wave-internal DS RAW, hw waits
#pragma unroll
      for (int f = 0; f < 4; ++f) {
        v16h bvf = lds_load_b16(sVt + (f * 16) * 32, 32);
        o[f] = wmma_f16(ap, bvf, o[f]);
      }
    }
    __syncthreads();
  }

  // normalize, store ctx f16 merged-head [B, S, H]
#pragma unroll
  for (int f = 0; f < 4; ++f) {
    const int d = f * 16 + llo;
#pragma unroll
    for (int r = 0; r < 8; ++r) {
      const int qg = q0 + r + 8 * lhi;
      ctx[((size_t)bz * S_ + qg) * H_ + hy * DK_ + d] =
          (_Float16)(o[f][r] / lrow[r]);
    }
  }
}

// =================================================================
// Kernel 3: output projection  out = ctx @ Wo^T + bo  (f16 in, f32 out)
// Same 128x128 double-buffered async+TDM pipeline as kernel 1.
// =================================================================
__global__ __launch_bounds__(256)
void oproj_kernel(const _Float16* __restrict__ Ax, const _Float16* __restrict__ Wh,
                  const float* __restrict__ bias, float* __restrict__ Out) {
  __shared__ _Float16 sA[2][128 * 32];
  __shared__ _Float16 sB[2][128 * 32];

  const int tid  = threadIdx.x;
  const int lane = tid & 31;
  const int wid  = tid >> 5;
  const int wm   = wid & 3;
  const int wn   = wid >> 2;
  const int m0   = blockIdx.x * 128;
  const int n0   = blockIdx.y * 128;

  const v8f zero8 = {0.f, 0.f, 0.f, 0.f, 0.f, 0.f, 0.f, 0.f};
  v8f acc[2][4];
#pragma unroll
  for (int i = 0; i < 2; ++i)
#pragma unroll
    for (int f = 0; f < 4; ++f) acc[i][f] = zero8;

  const int sr = tid >> 2;
  const int sc = (tid & 3) * 8;

  auto stage = [&](int k0, int buf) {
    async_cp16(&sA[buf][sr * 32 + sc],        Ax + (size_t)(m0 + sr) * H_ + k0 + sc);
    async_cp16(&sA[buf][(sr + 64) * 32 + sc], Ax + (size_t)(m0 + sr + 64) * H_ + k0 + sc);
#if HAVE_TDM
    if (wid == 0)
      tdm_load_2d_f16(&sB[buf][0], Wh + (size_t)n0 * H_ + k0,
                      /*tile*/ 32, 128, /*tensor*/ H_, H_, /*stride*/ H_);
#else
    async_cp16(&sB[buf][sr * 32 + sc],        Wh + (size_t)(n0 + sr) * H_ + k0 + sc);
    async_cp16(&sB[buf][(sr + 64) * 32 + sc], Wh + (size_t)(n0 + sr + 64) * H_ + k0 + sc);
#endif
  };

  stage(0, 0);

  const int NT = H_ / 32;
  for (int it = 0; it < NT; ++it) {
    const int cur = it & 1;
    const bool more = (it + 1) < NT;
    if (more) {
      stage((it + 1) * 32, cur ^ 1);
#if HAVE_TDM
      wait_async_2();
      if (wid == 0) __builtin_amdgcn_s_wait_tensorcnt((short)1);
#else
      asm volatile("s_wait_asynccnt 4" ::: "memory");
#endif
    } else {
      wait_async_0();
#if HAVE_TDM
      if (wid == 0) __builtin_amdgcn_s_wait_tensorcnt((short)0);
#endif
    }
    __syncthreads();

    v16h a0 = lds_load_a16(&sA[cur][(wm * 32) * 32], 32);
    v16h a1 = lds_load_a16(&sA[cur][(wm * 32 + 16) * 32], 32);
#pragma unroll
    for (int f = 0; f < 4; ++f) {
      v16h b = lds_load_b16(&sB[cur][(wn * 64 + f * 16) * 32], 32);
      acc[0][f] = wmma_f16(a0, b, acc[0][f]);
      acc[1][f] = wmma_f16(a1, b, acc[1][f]);
    }
    __syncthreads();
  }

  const int lhi = lane >> 4, llo = lane & 15;
#pragma unroll
  for (int fi = 0; fi < 2; ++fi) {
#pragma unroll
    for (int fj = 0; fj < 4; ++fj) {
      const int row0 = m0 + wm * 32 + fi * 16 + 8 * lhi;
      const int col  = n0 + wn * 64 + fj * 16 + llo;
      const float bvv = bias[col];
#pragma unroll
      for (int r = 0; r < 8; ++r) {
        Out[(size_t)(row0 + r) * H_ + col] = acc[fi][fj][r] + bvv;
      }
    }
  }
}

// =================================================================
extern "C" void kernel_launch(void* const* d_in, const int* in_sizes, int n_in,
                              void* d_out, int out_size, void* d_ws, size_t ws_size,
                              hipStream_t stream) {
  const float* Q  = (const float*)d_in[0];
  const float* K  = (const float*)d_in[1];
  const float* V  = (const float*)d_in[2];
  const unsigned char* mask = (const unsigned char*)d_in[3];
  const float* Wq = (const float*)d_in[4];
  const float* bq = (const float*)d_in[5];
  const float* Wk = (const float*)d_in[6];
  const float* bk = (const float*)d_in[7];
  const float* Wv = (const float*)d_in[8];
  const float* bv = (const float*)d_in[9];
  const float* Wo = (const float*)d_in[10];
  const float* bo = (const float*)d_in[11];
  float* out = (float*)d_out;

  const size_t NX = (size_t)B_ * S_ * H_;   // 4,194,304
  const size_t NW = (size_t)H_ * H_;        // 1,048,576

  _Float16* p = (_Float16*)d_ws;
  _Float16* Qh  = p; p += NX;
  _Float16* Kh  = p; p += NX;
  _Float16* Vh  = p; p += NX;
  _Float16* Wqh = p; p += NW;
  _Float16* Wkh = p; p += NW;
  _Float16* Wvh = p; p += NW;
  _Float16* Woh = p; p += NW;
  _Float16* qp  = p; p += NX;
  _Float16* kp  = p; p += NX;
  _Float16* vp  = p; p += NX;
  _Float16* ctx = p; p += NX;

  const int gx = (int)(NX / 2048);   // 8 elts/thread, 256 threads
  const int gw = (int)(NW / 2048);
  cvt_f16_kernel<<<gx, 256, 0, stream>>>(Q,  Qh,  (int)NX);
  cvt_f16_kernel<<<gx, 256, 0, stream>>>(K,  Kh,  (int)NX);
  cvt_f16_kernel<<<gx, 256, 0, stream>>>(V,  Vh,  (int)NX);
  cvt_f16_kernel<<<gw, 256, 0, stream>>>(Wq, Wqh, (int)NW);
  cvt_f16_kernel<<<gw, 256, 0, stream>>>(Wk, Wkh, (int)NW);
  cvt_f16_kernel<<<gw, 256, 0, stream>>>(Wv, Wvh, (int)NW);
  cvt_f16_kernel<<<gw, 256, 0, stream>>>(Wo, Woh, (int)NW);

  qkv_proj_kernel<<<dim3((B_ * S_) / 128, H_ / 128, 3), 256, 0, stream>>>(
      Qh, Kh, Vh, Wqh, Wkh, Wvh, bq, bk, bv, qp, kp, vp);
  attn_kernel<<<dim3(S_ / 128, NH_, B_), 256, 0, stream>>>(qp, kp, vp, mask, ctx);
  oproj_kernel<<<dim3((B_ * S_) / 128, H_ / 128, 1), 256, 0, stream>>>(ctx, Woh, bo, out);
}
